// Attention_28355374088877
// MI455X (gfx1250) — matrix operand order repaired and checked by reading, not compiled, not run on hardware
//
#include <hip/hip_runtime.h>

// Problem constants (from reference): B=4, F=T=2048, H=1024, N=16 heads, h=64
#define B_    4
#define SEQ   2048
#define HID   1024
#define NH    16
#define HS    64
#define NHID  1024   // NH*HS

typedef __attribute__((ext_vector_type(16))) __bf16 v16bf;
typedef __attribute__((ext_vector_type(8)))  __bf16 v8bf;
typedef __attribute__((ext_vector_type(8)))  float  v8f;
typedef __attribute__((ext_vector_type(4)))  int    v4i;

union Frag {          // one WMMA A/B operand (16 bf16 across 8 VGPRs)
  v16bf v;
  v8bf  h[2];
};

// f32 -> bf16 via native fptrunc (RNE); backend picks hw cvt on gfx1250
__device__ __forceinline__ unsigned short f2bf(float f) {
  __bf16 b = (__bf16)f;
  return __builtin_bit_cast(unsigned short, b);
}
__device__ __forceinline__ unsigned int f2bf_pk(float lo, float hi_) {
  return (unsigned int)f2bf(lo) | ((unsigned int)f2bf(hi_) << 16);
}

// ---- async global->LDS copy (CDNA5 GLOBAL_LOAD_ASYNC_TO_LDS_B128) ----
#if defined(__AMDGCN__) && __has_builtin(__builtin_amdgcn_global_load_async_to_lds_b128)
#define HAVE_ASYNC_LDS 1
#endif

__device__ __forceinline__ void cp_lds16(unsigned short* dst,
                                         const unsigned short* src) {
#ifdef HAVE_ASYNC_LDS
  __builtin_amdgcn_global_load_async_to_lds_b128(
      (__attribute__((address_space(1))) v4i*)src,
      (__attribute__((address_space(3))) v4i*)dst, 0, 0);
#else
  *(v8bf*)(void*)dst = *(const v8bf*)(const void*)src;
#endif
}
__device__ __forceinline__ void async_wait0() {
#ifdef HAVE_ASYNC_LDS
  asm volatile("s_wait_asynccnt 0" ::: "memory");
#endif
}

// ---------------------------------------------------------------------------
// Kernel 0: f32 activations -> bf16 copy (read once, reused 3x16 times later)
// ---------------------------------------------------------------------------
__global__ void __launch_bounds__(256)
xbf_kernel(const float* __restrict__ X, unsigned short* __restrict__ Y) {
  size_t i = ((size_t)blockIdx.x * 256 + threadIdx.x) * 8;
  float4 a = *(const float4*)&X[i];
  float4 b = *(const float4*)&X[i + 4];
  uint4 o;
  o.x = f2bf_pk(a.x, a.y);
  o.y = f2bf_pk(a.z, a.w);
  o.z = f2bf_pk(b.x, b.y);
  o.w = f2bf_pk(b.z, b.w);
  *(uint4*)(void*)&Y[i] = o;
}

// ---------------------------------------------------------------------------
// Kernel 1: W[H][N*h] f32  ->  Wt[N*h][H] bf16 (transposed, for B-fragments)
// ---------------------------------------------------------------------------
__global__ void __launch_bounds__(256)
wt_kernel(const float* __restrict__ W, unsigned short* __restrict__ Wt) {
  int k = blockIdx.y * 2;                          // 0,2,..HID-2
  int c = blockIdx.x * blockDim.x + threadIdx.x;   // 0..NHID-1
  unsigned int u = f2bf_pk(W[(size_t)k * NHID + c], W[(size_t)(k + 1) * NHID + c]);
  *(unsigned int*)(void*)&Wt[(size_t)c * HID + k] = u;
}

// ---------------------------------------------------------------------------
// Kernel 2: Y = X * W + b  via bf16 WMMA, f32 accumulate. K-step 64.
//   Xb:  [M=B*SEQ, HID] bf16
//   Wt:  [NHID, HID] bf16 (transposed weights)
//   vmode==0: Y bf16 row-major [M, NHID]                (Q, K)
//   vmode==1: Y bf16 transposed [B, NH, HS, SEQ]        (V)
// WG tile 128(M) x 64(N); 8 waves each compute 32x32 (2x2 WMMA tiles).
// Tiles staged via GLOBAL_LOAD_ASYNC_TO_LDS_B128 when available.
// ---------------------------------------------------------------------------
__global__ void __launch_bounds__(256)
proj_kernel(const unsigned short* __restrict__ Xb,
            const unsigned short* __restrict__ Wt,
            const float* __restrict__ bias,
            unsigned short* __restrict__ Y,
            int vmode) {
  __shared__ unsigned short Xs[128][72];  // 128 x 64 bf16 tile (+pad)
  __shared__ unsigned short Ws[64][72];   // 64 ncols x 64 k bf16 tile (+pad)

  const int tid  = threadIdx.x;
  const int lane = tid & 31;
  const int wave = tid >> 5;
  const int wm   = wave & 3;     // 4 M-slots of 32 rows
  const int wn   = wave >> 2;    // 2 N-slots of 32 cols
  const int r    = lane & 15;
  const int hi   = lane >> 4;
  const int mblock = blockIdx.y * 128;
  const int nblock = blockIdx.x * 64;

  v8f acc[2][2] = {};

  for (int k0 = 0; k0 < HID; k0 += 64) {
    // ---- stage X tile: 128 rows x 64 cols bf16, 4x16B per thread ----
    {
      int rr = tid >> 1;              // 0..127
      int cb = (tid & 1) * 32;        // 0 or 32
      const unsigned short* src = &Xb[(size_t)(mblock + rr) * HID + k0 + cb];
      cp_lds16(&Xs[rr][cb],      src);
      cp_lds16(&Xs[rr][cb + 8],  src + 8);
      cp_lds16(&Xs[rr][cb + 16], src + 16);
      cp_lds16(&Xs[rr][cb + 24], src + 24);
    }
    // ---- stage Wt tile: 64 ncols x 64 k bf16, 2x16B per thread ----
    {
      int rr = tid >> 2;              // 0..63
      int cb = (tid & 3) * 16;        // 0,16,32,48
      const unsigned short* src = &Wt[(size_t)(nblock + rr) * HID + k0 + cb];
      cp_lds16(&Ws[rr][cb],     src);
      cp_lds16(&Ws[rr][cb + 8], src + 8);
    }
    async_wait0();
    __syncthreads();

    // ---- 2 K-chunks of 32, A/B fragments in contiguous k order (k = 16*hi + s) ----
#pragma unroll
    for (int kc = 0; kc < 64; kc += 32) {
      Frag A[2], Bf[2];
#pragma unroll
      for (int i = 0; i < 2; ++i) {
        int rr = wm * 32 + i * 16 + r;
        A[i].h[0] = *(const v8bf*)(const void*)&Xs[rr][kc + hi * 16];
        A[i].h[1] = *(const v8bf*)(const void*)&Xs[rr][kc + hi * 16 + 8];
      }
#pragma unroll
      for (int j = 0; j < 2; ++j) {
        int cc = wn * 32 + j * 16 + r;
        Bf[j].h[0] = *(const v8bf*)(const void*)&Ws[cc][kc + hi * 16];
        Bf[j].h[1] = *(const v8bf*)(const void*)&Ws[cc][kc + hi * 16 + 8];
      }
#pragma unroll
      for (int i = 0; i < 2; ++i)
#pragma unroll
        for (int j = 0; j < 2; ++j)
          acc[i][j] = __builtin_amdgcn_wmma_f32_16x16x32_bf16(
              false, A[i].v, false, Bf[j].v, (short)0, acc[i][j], false, false);
    }
    __syncthreads();
  }

  // ---- epilogue: bias + store ----
#pragma unroll
  for (int i = 0; i < 2; ++i) {
#pragma unroll
    for (int j = 0; j < 2; ++j) {
      int c  = nblock + wn * 32 + j * 16 + r;        // output column
      float bc = bias[c];
      int row0 = mblock + wm * 32 + i * 16 + hi * 8;  // first of 8 rows
      if (vmode == 0) {
#pragma unroll
        for (int v = 0; v < 8; ++v)
          Y[(size_t)(row0 + v) * NHID + c] = f2bf(acc[i][j][v] + bc);
      } else {
        int b  = row0 / SEQ, t = row0 % SEQ;
        int n  = c / HS,    hc = c % HS;
        uint4 o;
        o.x = f2bf_pk(acc[i][j][0] + bc, acc[i][j][1] + bc);
        o.y = f2bf_pk(acc[i][j][2] + bc, acc[i][j][3] + bc);
        o.z = f2bf_pk(acc[i][j][4] + bc, acc[i][j][5] + bc);
        o.w = f2bf_pk(acc[i][j][6] + bc, acc[i][j][7] + bc);
        *(uint4*)(void*)&Y[(((size_t)b * NH + n) * HS + hc) * SEQ + t] = o;
      }
    }
  }
}

// ---------------------------------------------------------------------------
// Kernel 3: flash attention. One WG = (b, n, 128-row F block), 8 waves,
// wave owns 16 F rows; streams T in steps of 32 with online softmax.
//   Q  [B,SEQ,NH,HS] bf16, K [B,SEQ,NH,HS] bf16, Vt [B,NH,HS,SEQ] bf16
//   mask [B,F,T] f32, out [B,F,NH,HS] f32
// ---------------------------------------------------------------------------
__global__ void __launch_bounds__(256)
attn_kernel(const unsigned short* __restrict__ Q,
            const unsigned short* __restrict__ K,
            const unsigned short* __restrict__ Vt,
            const float* __restrict__ mask,
            float* __restrict__ out) {
  __shared__ unsigned short Plds[8][16][32];  // per-wave prob staging

  const int tid  = threadIdx.x;
  const int lane = tid & 31;
  const int wave = tid >> 5;
  const int r    = lane & 15;
  const int hi   = lane >> 4;

  const int FB   = SEQ / 128;                  // 16 F-blocks
  const int wg   = blockIdx.x;
  const int fblk = wg % FB;
  const int n    = (wg / FB) % NH;
  const int b    = wg / (FB * NH);
  const int fbase = fblk * 128 + wave * 16;

  // Q A-fragments (row = f, K-dim = h), loaded once
  Frag Qa[2];
  {
    const unsigned short* qp =
        Q + (((size_t)b * SEQ + fbase + r) * NH + n) * HS;
#pragma unroll
    for (int c = 0; c < 2; ++c) {
      Qa[c].h[0] = *(const v8bf*)(const void*)(qp + c * 32 + hi * 16);
      Qa[c].h[1] = *(const v8bf*)(const void*)(qp + c * 32 + hi * 16 + 8);
    }
  }

  v8f O[4] = {};
  float m[8], l[8];
#pragma unroll
  for (int v = 0; v < 8; ++v) { m[v] = -1e30f; l[v] = 0.0f; }

  const float scale = 0.125f;  // 1/sqrt(64)
  const unsigned short* kbase = K + (size_t)b * SEQ * NHID + (size_t)n * HS;
  const unsigned short* vbase = Vt + ((size_t)b * NH + n) * (size_t)HS * SEQ;
  const float* mrow = mask + (size_t)b * SEQ * SEQ;

  for (int t0 = 0; t0 < SEQ; t0 += 32) {
    if (t0 + 32 < SEQ)  // pull next K tile toward the caches
      __builtin_prefetch(kbase + (size_t)(t0 + 32 + r) * NHID, 0, 1);

    // ---- scores: two 16x16 tiles over 32 T-columns, K-dim h=64 ----
    v8f s0 = {}, s1 = {};
#pragma unroll
    for (int c = 0; c < 2; ++c) {
      Frag kb0, kb1;
      const unsigned short* kp0 =
          kbase + (size_t)(t0 + r) * NHID + c * 32 + hi * 16;
      const unsigned short* kp1 =
          kbase + (size_t)(t0 + 16 + r) * NHID + c * 32 + hi * 16;
      kb0.h[0] = *(const v8bf*)(const void*)kp0;
      kb0.h[1] = *(const v8bf*)(const void*)(kp0 + 8);
      kb1.h[0] = *(const v8bf*)(const void*)kp1;
      kb1.h[1] = *(const v8bf*)(const void*)(kp1 + 8);
      s0 = __builtin_amdgcn_wmma_f32_16x16x32_bf16(
          false, Qa[c].v, false, kb0.v, (short)0, s0, false, false);
      s1 = __builtin_amdgcn_wmma_f32_16x16x32_bf16(
          false, Qa[c].v, false, kb1.v, (short)0, s1, false, false);
    }

    // ---- scale + mask adder + online softmax (row = hi*8+v) ----
#pragma unroll
    for (int v = 0; v < 8; ++v) {
      int f = fbase + hi * 8 + v;
      const float* mr = mrow + (size_t)f * SEQ + t0;
      float a0 = s0[v] * scale + (1.0f - mr[r])      * -10000.0f;
      float a1 = s1[v] * scale + (1.0f - mr[16 + r]) * -10000.0f;

      float tm = fmaxf(a0, a1);
#pragma unroll
      for (int off = 1; off < 16; off <<= 1)
        tm = fmaxf(tm, __shfl_xor(tm, off, 32));
      float mn   = fmaxf(m[v], tm);
      float corr = __expf(m[v] - mn);
      float p0   = __expf(a0 - mn);
      float p1   = __expf(a1 - mn);
      float rs   = p0 + p1;
#pragma unroll
      for (int off = 1; off < 16; off <<= 1)
        rs += __shfl_xor(rs, off, 32);
      l[v] = l[v] * corr + rs;
      m[v] = mn;
#pragma unroll
      for (int ht = 0; ht < 4; ++ht) O[ht][v] *= corr;

      Plds[wave][hi * 8 + v][r]      = f2bf(p0);
      Plds[wave][hi * 8 + v][16 + r] = f2bf(p1);
    }
    __syncthreads();  // order C-layout stores before A-fragment reads

    // ---- P (A-frag 16x32 over t) x V (B-frag) -> O over h=64 ----
    Frag Pa;
    Pa.h[0] = *(const v8bf*)(const void*)&Plds[wave][r][hi * 16];
    Pa.h[1] = *(const v8bf*)(const void*)&Plds[wave][r][hi * 16 + 8];
#pragma unroll
    for (int ht = 0; ht < 4; ++ht) {
      Frag vb;
      const unsigned short* vp =
          vbase + (size_t)(ht * 16 + r) * SEQ + t0 + hi * 16;
      vb.h[0] = *(const v8bf*)(const void*)vp;
      vb.h[1] = *(const v8bf*)(const void*)(vp + 8);
      O[ht] = __builtin_amdgcn_wmma_f32_16x16x32_bf16(
          false, Pa.v, false, vb.v, (short)0, O[ht], false, false);
    }
  }

  // ---- normalize + store out[b][f][n][h] f32 ----
#pragma unroll
  for (int ht = 0; ht < 4; ++ht) {
#pragma unroll
    for (int v = 0; v < 8; ++v) {
      int f = fbase + hi * 8 + v;
      out[(((size_t)b * SEQ + f) * NH + n) * HS + ht * 16 + r] =
          O[ht][v] / l[v];
    }
  }
}

// ---------------------------------------------------------------------------
extern "C" void kernel_launch(void* const* d_in, const int* in_sizes, int n_in,
                              void* d_out, int out_size, void* d_ws,
                              size_t ws_size, hipStream_t stream) {
  const float* from = (const float*)d_in[0];   // [B,F,H]
  const float* to   = (const float*)d_in[1];   // [B,T,H]
  const float* mask = (const float*)d_in[2];   // [B,F,T]
  const float* Wq   = (const float*)d_in[3];
  const float* bq   = (const float*)d_in[4];
  const float* Wk   = (const float*)d_in[5];
  const float* bk   = (const float*)d_in[6];
  const float* Wv   = (const float*)d_in[7];
  const float* bv   = (const float*)d_in[8];
  float* out = (float*)d_out;

  // workspace layout (bf16 halves), total ~86 MiB
  unsigned short* Wtq = (unsigned short*)d_ws;
  unsigned short* Wtk = Wtq + (size_t)NHID * HID;
  unsigned short* Wtv = Wtk + (size_t)NHID * HID;
  unsigned short* Qb  = Wtv + (size_t)NHID * HID;
  unsigned short* Kb  = Qb + (size_t)B_ * SEQ * NHID;
  unsigned short* Vt  = Kb + (size_t)B_ * SEQ * NHID;
  unsigned short* Fbf = Vt + (size_t)B_ * SEQ * NHID;
  unsigned short* Tbf = Fbf + (size_t)B_ * SEQ * HID;

  dim3 tb(256);

  xbf_kernel<<<dim3((B_ * SEQ * HID) / (256 * 8)), tb, 0, stream>>>(from, Fbf);
  xbf_kernel<<<dim3((B_ * SEQ * HID) / (256 * 8)), tb, 0, stream>>>(to, Tbf);

  dim3 tg(NHID / 256, HID / 2);
  wt_kernel<<<tg, tb, 0, stream>>>(Wq, Wtq);
  wt_kernel<<<tg, tb, 0, stream>>>(Wk, Wtk);
  wt_kernel<<<tg, tb, 0, stream>>>(Wv, Wtv);

  dim3 pg(NHID / 64, (B_ * SEQ) / 128);
  proj_kernel<<<pg, tb, 0, stream>>>(Fbf, Wtq, bq, Qb, 0);
  proj_kernel<<<pg, tb, 0, stream>>>(Tbf, Wtk, bk, Kb, 0);
  proj_kernel<<<pg, tb, 0, stream>>>(Tbf, Wtv, bv, Vt, 1);

  dim3 ag(B_ * NH * (SEQ / 128));
  attn_kernel<<<ag, tb, 0, stream>>>(Qb, Kb, Vt, mask, out);
}
